// FeatureFlipFusion_40596030882008
// MI455X (gfx1250) — matrix operand order repaired and checked by reading, not compiled, not run on hardware
//
#include <hip/hip_runtime.h>
#include <hip/hip_bf16.h>
#include <math.h>

// ---- CDNA5 WMMA types (wave32) ----
typedef __attribute__((ext_vector_type(16))) _Float16 v16h;
typedef __attribute__((ext_vector_type(8)))  float    v8f;

#define C_CH   128
#define H_DIM  64
#define W_DIM  160
#define BATCH  4
#define HW     (H_DIM * W_DIM)   // 10240
#define NTOT   (BATCH * HW)      // 40960
#define EPSB   1e-5f
#define KOFF   2304              // 256 * 9  (offset conv im2col K)
#define KDCN   1152              // 128 * 9  (dcn GEMM K)

// ---------------------------------------------------------------------------
// Fragment-order LDS indexing for V_WMMA_F32_16X16X32_F16 (ISA 7.12.2):
//  A (16x32 f16, MxK): lanes 0-15 -> M=lane, VGPR0..3 = K 0..7, VGPR4..7 = K 16..23
//                      lanes 16-31 -> M=lane-16, VGPR0..3 = K 8..15, VGPR4..7 = K 24..31
//  B (32x16 f16, KxN): lanes 0-15 -> N=lane, halves 0..7 = K 0..7, halves 8..15 = K 16..23
//                      lanes 16-31 -> N=lane-16, halves 0..7 = K 8..15, halves 8..15 = K 24..31
// We stage tiles into LDS so that each lane's 16 halves are contiguous
// (lane*16 + elem), enabling a single 32B ds read per fragment per lane.
// ---------------------------------------------------------------------------
__device__ __forceinline__ int frag_a_idx(int m, int k) {
  int half, i;
  if (k < 16) { half = k >> 3;        i = k & 7; }
  else        { int kk = k - 16; half = kk >> 3; i = 8 + (kk & 7); }
  int lane = (half << 4) | m;
  return lane * 16 + i;
}
__device__ __forceinline__ int frag_b_idx(int k, int n) {
  int lane = (((k >> 3) & 1) << 4) | n;   // half = (k>>3)&1
  int i = (k & 7) | ((k & 16) >> 1);
  return lane * 16 + i;
}

__device__ __forceinline__ v8f wmma_f16(v16h a, v16h b, v8f c) {
  return __builtin_amdgcn_wmma_f32_16x16x32_f16(
      /*neg_a=*/false, a, /*neg_b=*/false, b,
      /*c_mod=*/(short)0, c, /*reuse_a=*/false, /*reuse_b=*/false);
}

// bilinear tap on the x-flipped feature map (flip folded into the index)
__device__ __forceinline__ float tap_flip(const float* __restrict__ base,
                                          int y, int x, float w) {
  if (y < 0 || y >= H_DIM || x < 0 || x >= W_DIM) return 0.0f;
  return base[y * W_DIM + (W_DIM - 1 - x)] * w;
}

// ===========================================================================
// Kernel A: feat = BN1(feature x W1 + b1)   GEMM 128 x 128 x 40960
// block = 256 thr (8 waves). wave w owns M-strip [16w,16w+16), N tile = 64.
// ===========================================================================
__global__ __launch_bounds__(256)
void conv1x1_bn_kernel(const float* __restrict__ feature,
                       const float* __restrict__ w1,
                       const float* __restrict__ b1,
                       const float* __restrict__ g1,
                       const float* __restrict__ be1,
                       const float* __restrict__ m1,
                       const float* __restrict__ v1,
                       float* __restrict__ feat_out) {
  __shared__ __align__(32) _Float16 ldsA[8 * 512];   // 128 x 32 (8 strips)
  __shared__ __align__(32) _Float16 ldsB[4 * 512];   // 32 x 64  (4 N tiles)
  const int tid  = threadIdx.x;
  const int wave = tid >> 5;
  const int lane = tid & 31;
  const int pixbase = blockIdx.x * 64;
  const int b   = pixbase / HW;
  const int hw0 = pixbase - b * HW;     // 64-pixel tile stays in one batch image

  v8f acc[4];
  #pragma unroll
  for (int t = 0; t < 4; ++t) acc[t] = (v8f){};

  for (int kb = 0; kb < C_CH; kb += 32) {
    // --- stage A: W1[o, kb..kb+32) as f16 in fragment order ---
    for (int e = tid; e < 128 * 32; e += 256) {
      int o = e >> 5, k = e & 31;
      ldsA[(o >> 4) * 512 + frag_a_idx(o & 15, k)] =
          (_Float16)w1[o * C_CH + kb + k];
    }
    // --- stage B: feature[b, kb+k, hw0+nl] ---
    for (int e = tid; e < 32 * 64; e += 256) {
      int nl = e & 63, k = e >> 6;
      float x = feature[(size_t)b * C_CH * HW + (size_t)(kb + k) * HW + hw0 + nl];
      ldsB[(nl >> 4) * 512 + frag_b_idx(k, nl & 15)] = (_Float16)x;
    }
    // prefetch next K-step's activation rows into cache (global_prefetch_b8)
    if (kb + 32 < C_CH) {
      const float* p = feature + (size_t)b * C_CH * HW +
                       (size_t)(kb + 32 + (tid >> 3)) * HW + hw0;
      __builtin_prefetch(p, 0, 1);
    }
    __syncthreads();
    v16h a = *(const v16h*)(ldsA + wave * 512 + lane * 16);
    #pragma unroll
    for (int t = 0; t < 4; ++t) {
      v16h bf = *(const v16h*)(ldsB + t * 512 + lane * 16);
      acc[t] = wmma_f16(a, bf, acc[t]);
    }
    __syncthreads();
  }

  // epilogue: + b1, BN1, store fp32 feat[o][n]
  const int half = lane >> 4, nl = lane & 15;
  #pragma unroll
  for (int t = 0; t < 4; ++t) {
    #pragma unroll
    for (int v = 0; v < 8; ++v) {
      int o = wave * 16 + v + 8 * half;
      int n = pixbase + t * 16 + nl;
      float inv = g1[o] * rsqrtf(v1[o] + EPSB);
      float val = acc[t][v] + b1[o];
      val = (val - m1[o]) * inv + be1[o];
      feat_out[(size_t)o * NTOT + n] = val;
    }
  }
}

// ===========================================================================
// Kernel B: offset conv 3x3 over concat[flipped, feat] -> py/px/mask fields.
// im2col GEMM 32(pad27) x 2304 x 40960. 8 waves: strip = w&1, N tile = w>>1.
// ===========================================================================
__global__ __launch_bounds__(256)
void offset_conv_kernel(const float* __restrict__ feature,
                        const float* __restrict__ feat,
                        const float* __restrict__ off_w,
                        const float* __restrict__ off_b,
                        float* __restrict__ PY,
                        float* __restrict__ PX,
                        float* __restrict__ MASK) {
  __shared__ __align__(32) _Float16 ldsA[2 * 512];   // 32 x 32
  __shared__ __align__(32) _Float16 ldsB[4 * 512];   // 32 x 64
  const int tid  = threadIdx.x;
  const int wave = tid >> 5;
  const int lane = tid & 31;
  const int strip = wave & 1;
  const int ptile = wave >> 1;
  const int pixbase = blockIdx.x * 64;
  const int b   = pixbase / HW;
  const int hw0 = pixbase - b * HW;

  v8f acc = (v8f){};

  for (int kb = 0; kb < KOFF; kb += 32) {
    // --- stage A: off_w[o, cc, tap] with k = tap*256 + cc; pad rows 27..31 ---
    for (int e = tid; e < 32 * 32; e += 256) {
      int o = e >> 5, k = e & 31;
      int kg = kb + k;
      int t = kg >> 8, cc = kg & 255;
      float wv = (o < 27) ? off_w[(o * 256 + cc) * 9 + t] : 0.0f;
      ldsA[(o >> 4) * 512 + frag_a_idx(o & 15, k)] = (_Float16)wv;
    }
    // --- stage B: im2col of [flipped || feat], zero pad at borders ---
    for (int e = tid; e < 32 * 64; e += 256) {
      int nl = e & 63, k = e >> 6;
      int kg = kb + k;
      int t = kg >> 8, cc = kg & 255;
      int dy = t / 3 - 1, dx = t - (t / 3) * 3 - 1;
      int hw = hw0 + nl;
      int h  = hw / W_DIM;
      int w_ = hw - h * W_DIM;
      int y = h + dy, x = w_ + dx;
      float xv = 0.0f;
      if (y >= 0 && y < H_DIM && x >= 0 && x < W_DIM) {
        if (cc < C_CH)
          xv = feature[(size_t)b * C_CH * HW + (size_t)cc * HW +
                       y * W_DIM + (W_DIM - 1 - x)];            // flipped
        else
          xv = feat[(size_t)(cc - C_CH) * NTOT + b * HW + y * W_DIM + x];
      }
      ldsB[(nl >> 4) * 512 + frag_b_idx(k, nl & 15)] = (_Float16)xv;
    }
    __syncthreads();
    v16h a  = *(const v16h*)(ldsA + strip * 512 + lane * 16);
    v16h bf = *(const v16h*)(ldsB + ptile * 512 + lane * 16);
    acc = wmma_f16(a, bf, acc);
    __syncthreads();
  }

  // epilogue: split 27 channels -> py (+h+ky), px (+w+kx), sigmoid(mask)
  const int half = lane >> 4, nl = lane & 15;
  #pragma unroll
  for (int v = 0; v < 8; ++v) {
    int o = strip * 16 + v + 8 * half;
    if (o >= 27) continue;
    int n  = pixbase + ptile * 16 + nl;
    int hw = n % HW;
    int h  = hw / W_DIM;
    int w_ = hw - h * W_DIM;
    float val = acc[v] + off_b[o];
    if (o < 9) {
      PY[(size_t)o * NTOT + n] = val + (float)h + (float)(o / 3 - 1);
    } else if (o < 18) {
      int k = o - 9;
      PX[(size_t)k * NTOT + n] = val + (float)w_ + (float)(k % 3 - 1);
    } else {
      MASK[(size_t)(o - 18) * NTOT + n] = 1.0f / (1.0f + __expf(-val));
    }
  }
}

// ===========================================================================
// Kernel D: deformable GEMM 128 x 1152 x 40960 with the bilinear gather
// fused into B-tile staging (no 94MB sampled intermediate).
// Epilogue: +dcn_b, BN2, relu(feat + dout), scatter NCHW.
// ===========================================================================
__global__ __launch_bounds__(256)
void dcn_gemm_kernel(const float* __restrict__ feature,
                     const float* __restrict__ feat,
                     const float* __restrict__ dcn_w,
                     const float* __restrict__ dcn_b,
                     const float* __restrict__ g2,
                     const float* __restrict__ be2,
                     const float* __restrict__ m2,
                     const float* __restrict__ v2,
                     const float* __restrict__ PY,
                     const float* __restrict__ PX,
                     const float* __restrict__ MASK,
                     float* __restrict__ out) {
  __shared__ __align__(32) _Float16 ldsA[8 * 512];   // 128 x 32
  __shared__ __align__(32) _Float16 ldsB[512];       // 32 x 16
  const int tid  = threadIdx.x;
  const int wave = tid >> 5;
  const int lane = tid & 31;
  const int pixbase = blockIdx.x * 16;
  const int b = pixbase / HW;

  v8f acc = (v8f){};

  for (int kb = 0; kb < KDCN; kb += 32) {
    // --- stage A: dcn_w flat (o, c*9+k) is exactly row-major, k-contig ---
    for (int e = tid; e < 128 * 32; e += 256) {
      int o = e >> 5, k = e & 31;
      ldsA[(o >> 4) * 512 + frag_a_idx(o & 15, k)] =
          (_Float16)dcn_w[(size_t)o * KDCN + kb + k];
    }
    // --- stage B: masked bilinear sample of flipped feature, r = c*9 + kk ---
    for (int e = tid; e < 512; e += 256) {
      int nl = e & 15, k = e >> 4;
      int r  = kb + k;
      int c  = r / 9;
      int kk = r - c * 9;
      int n  = pixbase + nl;
      float py = PY[(size_t)kk * NTOT + n];
      float px = PX[(size_t)kk * NTOT + n];
      float mk = MASK[(size_t)kk * NTOT + n];
      float y0f = floorf(py), x0f = floorf(px);
      int   y0 = (int)y0f,   x0 = (int)x0f;
      float wy = py - y0f,   wx = px - x0f;
      const float* base = feature + (size_t)b * C_CH * HW + (size_t)c * HW;
      float s = 0.0f;
      s += tap_flip(base, y0,     x0,     (1.0f - wy) * (1.0f - wx));
      s += tap_flip(base, y0,     x0 + 1, (1.0f - wy) * wx);
      s += tap_flip(base, y0 + 1, x0,     wy * (1.0f - wx));
      s += tap_flip(base, y0 + 1, x0 + 1, wy * wx);
      ldsB[frag_b_idx(k, nl)] = (_Float16)(s * mk);
    }
    __syncthreads();
    v16h a  = *(const v16h*)(ldsA + wave * 512 + lane * 16);
    v16h bf = *(const v16h*)(ldsB + lane * 16);
    acc = wmma_f16(a, bf, acc);
    __syncthreads();
  }

  // epilogue: bias, BN2, residual ReLU, NCHW store
  const int half = lane >> 4, nl = lane & 15;
  #pragma unroll
  for (int v = 0; v < 8; ++v) {
    int o = wave * 16 + v + 8 * half;
    int n = pixbase + nl;
    float dout = acc[v] + dcn_b[o];
    float inv  = g2[o] * rsqrtf(v2[o] + EPSB);
    dout = (dout - m2[o]) * inv + be2[o];
    float r = feat[(size_t)o * NTOT + n] + dout;
    r = fmaxf(r, 0.0f);
    int bb = n / HW;
    int hw = n - bb * HW;
    out[((size_t)bb * C_CH + o) * HW + hw] = r;
  }
}

// ===========================================================================
extern "C" void kernel_launch(void* const* d_in, const int* in_sizes, int n_in,
                              void* d_out, int out_size, void* d_ws, size_t ws_size,
                              hipStream_t stream) {
  const float* feature = (const float*)d_in[0];
  const float* w1      = (const float*)d_in[1];
  const float* b1      = (const float*)d_in[2];
  const float* g1      = (const float*)d_in[3];
  const float* be1     = (const float*)d_in[4];
  const float* m1      = (const float*)d_in[5];
  const float* v1      = (const float*)d_in[6];
  const float* off_w   = (const float*)d_in[7];
  const float* off_b   = (const float*)d_in[8];
  const float* dcn_w   = (const float*)d_in[9];
  const float* dcn_b   = (const float*)d_in[10];
  const float* g2      = (const float*)d_in[11];
  const float* be2     = (const float*)d_in[12];
  const float* m2      = (const float*)d_in[13];
  const float* v2      = (const float*)d_in[14];
  float* out = (float*)d_out;

  float* feat = (float*)d_ws;                         // 128 x 40960 fp32
  float* PY   = feat + (size_t)C_CH * NTOT;           // 9 x 40960
  float* PX   = PY   + (size_t)9 * NTOT;
  float* MASK = PX   + (size_t)9 * NTOT;

  conv1x1_bn_kernel<<<NTOT / 64, 256, 0, stream>>>(
      feature, w1, b1, g1, be1, m1, v1, feat);

  offset_conv_kernel<<<NTOT / 64, 256, 0, stream>>>(
      feature, feat, off_w, off_b, PY, PX, MASK);

  dcn_gemm_kernel<<<NTOT / 16, 256, 0, stream>>>(
      feature, feat, dcn_w, dcn_b, g2, be2, m2, v2, PY, PX, MASK, out);
}
